// CategoryInterestAttention_33818572488827
// MI455X (gfx1250) — matrix operand order repaired and checked by reading, not compiled, not run on hardware
//
#include <hip/hip_runtime.h>

#define B_ 512
#define T_ 256
#define D_ 128
#define C_ 64
#define H_ 4
#define HD_ 32
#define L_ 2
#define F_ 512

typedef __attribute__((ext_vector_type(16))) __bf16 v16bf;
typedef __attribute__((ext_vector_type(2)))  __bf16 v2bf;
typedef __attribute__((ext_vector_type(2)))  float  v2f;
typedef __attribute__((ext_vector_type(8)))  float  v8f;

union Frag {
    v16bf    v;
    int4     q[2];
    unsigned u[8];
};
union Pk2 { v2bf v; unsigned u; unsigned short s[2]; };

// f32 pair -> packed bf16 (RNE). fptrunc f32->bf16 is legal on all AMDGPU
// targets; on gfx1250 it selects the native cvt, otherwise expands to the
// integer RNE sequence.
__device__ __forceinline__ unsigned pack2_bf16(float a, float b) {
    v2f f; f.x = a; f.y = b;
    Pk2 p;
    p.v = __builtin_convertvector(f, v2bf);
    return p.u;
}
__device__ __forceinline__ unsigned short bf16_1(float a) {
    Pk2 p;
    p.v.x = (__bf16)a;
    return p.s[0];
}

// A-matrix fragment (16x32 bf16, M x K) from row-major bf16 buffer.
// Layout: lanes 0-15 hold K = k0..k0+15 contiguous; lanes 16-31 K = k0+16..k0+31.
__device__ __forceinline__ v16bf fragA_bf(const unsigned short* buf, int ld, int row0, int k0, int lane) {
    int r = row0 + (lane & 15);
    const unsigned short* p = buf + r * ld + k0 + ((lane >> 4) << 4);
    Frag f;
    f.q[0] = *(const int4*)(p);
    f.q[1] = *(const int4*)(p + 8);
    return f.v;
}

// B-matrix fragment (32x16 bf16, K x N) from an N x K row-major bf16 buffer.
// Layout: lanes 0-15 hold K = k0..k0+15 contiguous; lanes 16-31 K = k0+16..k0+31.
__device__ __forceinline__ v16bf fragB_bf(const unsigned short* buf, int ld, int row0, int k0, int lane) {
    int r = row0 + (lane & 15);
    const unsigned short* p = buf + r * ld + k0 + ((lane >> 4) << 4);
    Frag f;
    f.q[0] = *(const int4*)(p);
    f.q[1] = *(const int4*)(p + 8);
    return f.v;
}

// B fragment from N x K row-major f32 buffer (global weights), cvt->bf16.
__device__ __forceinline__ v16bf fragB_f32(const float* buf, int ld, int row0, int k0, int lane) {
    int r = row0 + (lane & 15);
    const float* p = buf + r * ld + k0 + ((lane >> 4) << 4);
    float4 a0 = *(const float4*)(p);
    float4 a1 = *(const float4*)(p + 4);
    float4 a2 = *(const float4*)(p + 8);
    float4 a3 = *(const float4*)(p + 12);
    Frag f;
    f.u[0] = pack2_bf16(a0.x, a0.y); f.u[1] = pack2_bf16(a0.z, a0.w);
    f.u[2] = pack2_bf16(a1.x, a1.y); f.u[3] = pack2_bf16(a1.z, a1.w);
    f.u[4] = pack2_bf16(a2.x, a2.y); f.u[5] = pack2_bf16(a2.z, a2.w);
    f.u[6] = pack2_bf16(a3.x, a3.y); f.u[7] = pack2_bf16(a3.z, a3.w);
    return f.v;
}

__device__ __forceinline__ v8f wmma_bf16(v16bf a, v16bf b, v8f c) {
    return __builtin_amdgcn_wmma_f32_16x16x32_bf16(false, a, false, b, (short)0, c, false, false);
}

// ---- shared memory layout (bytes) ----
#define OFF_X    0         // C*D f32              : 32768
#define OFF_K    32768     // T*D bf16 (k)         : 65536
#define OFF_VT   98304     // D*T bf16 (v^T)       : 65536
#define OFF_U0   163840    // union 65536: kn bf16[T][D] | scores f32[C][T] | h1 bf16[C][F]
#define OFF_U1   229376    // union 16384: qn/ctx/xn bf16 [C][D]
#define OFF_Q    245760    // q bf16 [C][D]        : 16384
#define OFF_AB   262144    // attn bf16 [C][T]     : 32768
#define OFF_TAIL 294912    // small arrays (6144)
#define SMEM_TOTAL 301056

extern "C" __global__ void __launch_bounds__(256)
cia_kernel(const float* __restrict__ tgt, const float* __restrict__ seq,
           const int* __restrict__ cats, const unsigned char* __restrict__ maskp,
           const float* __restrict__ wqkv, const float* __restrict__ bqkv,
           const float* __restrict__ wo,   const float* __restrict__ bo,
           const float* __restrict__ ln1g, const float* __restrict__ ln1b,
           const float* __restrict__ ln2g, const float* __restrict__ ln2b,
           const float* __restrict__ w1,   const float* __restrict__ b1,
           const float* __restrict__ w2,   const float* __restrict__ b2,
           float* __restrict__ out)
{
    extern __shared__ char smem[];
    float*          xbuf   = (float*)(smem + OFF_X);             // [C][D] f32
    unsigned short* kbuf   = (unsigned short*)(smem + OFF_K);    // [T][D] bf16
    unsigned short* vTbuf  = (unsigned short*)(smem + OFF_VT);   // [D][T] bf16
    unsigned short* knbuf  = (unsigned short*)(smem + OFF_U0);   // [T][D] bf16
    float*          sbuf   = (float*)(smem + OFF_U0);            // [C][T] f32 (union)
    unsigned short* h1buf  = (unsigned short*)(smem + OFF_U0);   // [C][F] bf16 (union)
    unsigned short* u1     = (unsigned short*)(smem + OFF_U1);   // [C][D] bf16 qn/ctx/xn
    unsigned short* qbuf   = (unsigned short*)(smem + OFF_Q);    // [C][D] bf16
    unsigned short* abuf   = (unsigned short*)(smem + OFF_AB);   // [C][T] bf16 attn
    int*   catsl   = (int*)(smem + OFF_TAIL);          // [T]
    int*   maskl   = (int*)(smem + OFF_TAIL + 1024);   // [T]
    int*   lastpos = (int*)(smem + OFF_TAIL + 2048);   // [64]
    int*   present = (int*)(smem + OFF_TAIL + 2304);   // [64]
    int*   catval  = (int*)(smem + OFF_TAIL + 2560);   // [64]
    int*   qidx    = (int*)(smem + OFF_TAIL + 2816);   // [64]
    float* logits  = (float*)(smem + OFF_TAIL + 3072); // [64]
    float* wsm     = (float*)(smem + OFF_TAIL + 3328); // [64]
    float* redm    = (float*)(smem + OFF_TAIL + 3584); // [4][64]
    float* reds    = (float*)(smem + OFF_TAIL + 4608); // [4][64]
    int*   nuniq   = (int*)(smem + OFF_TAIL + 5632);

    const int tid  = threadIdx.x;
    const int b    = blockIdx.x;
    const int lane = tid & 31;
    const int wave = tid >> 5;
    const int nlo  = lane & 15;
    const int half = lane >> 4;

    // ---------- Phase 0: unique categories, last-match positions, gather x ----------
    if (tid < 64) { lastpos[tid] = -1; present[tid] = 0; }
    __syncthreads();
    {
        int t  = tid;                       // blockDim == T
        int cv = cats[(long)b * T_ + t] & 63;
        int mk = maskp[(long)b * T_ + t] ? 1 : 0;
        catsl[t] = cv;
        maskl[t] = mk;
        if (mk) { atomicMax(&lastpos[cv], t); atomicOr(&present[cv], 1); }
    }
    __syncthreads();
    if (tid == 0) {
        int n = 0;
        for (int c2 = 0; c2 < 64; ++c2) if (present[c2]) catval[n++] = c2;
        *nuniq = n;
        for (int i = n; i < C_; ++i) catval[i] = 0;   // reference pads with cat 0
    }
    __syncthreads();
    if (tid < C_) {
        int lp = lastpos[catval[tid]];
        qidx[tid] = (lp < 0) ? 0 : lp;
    }
    __syncthreads();
    for (int i = tid; i < (C_ * D_) / 4; i += 256) {
        int c = i >> 5, d4 = i & 31;
        *(float4*)(xbuf + c * D_ + d4 * 4) =
            *(const float4*)(seq + ((long)b * T_ + qidx[c]) * (long)D_ + d4 * 4);
    }
    __syncthreads();

    // ---------- Transformer layers ----------
    for (int l = 0; l < L_; ++l) {
        const float* Wq  = wqkv + (long)l * 3 * D_ * D_;
        const float* Wk  = Wq + D_ * D_;
        const float* Wv  = Wk + D_ * D_;
        const float* bq  = bqkv + (long)l * 3 * D_;
        const float* bk  = bq + D_;
        const float* bv  = bk + D_;
        const float* g1  = ln1g + (long)l * D_;
        const float* be1 = ln1b + (long)l * D_;
        const float* W1_ = w1 + (long)l * F_ * D_;
        const float* b1_ = b1 + (long)l * F_;
        const float* W2_ = w2 + (long)l * D_ * F_;
        const float* b2_ = b2 + (long)l * D_;
        const float* Wo_ = wo + (long)l * D_ * D_;
        const float* bo_ = bo + (long)l * D_;

        // Prefetch this layer's weights (global_prefetch_b8).
        for (int i = tid; i < (F_ * D_) / 32; i += 256) {
            __builtin_prefetch(W1_ + i * 32, 0, 1);
            __builtin_prefetch(W2_ + i * 32, 0, 1);
        }
        for (int i = tid; i < (3 * D_ * D_) / 32; i += 256)
            __builtin_prefetch(Wq + i * 32, 0, 1);

        // kn = LN1(seq) -> bf16, one thread per sequence row (float4 vectorized)
        {
            const float* row = seq + ((long)b * T_ + tid) * (long)D_;
            float s = 0.f, ss = 0.f;
            for (int d = 0; d < D_; d += 4) {
                float4 v = *(const float4*)(row + d);
                s  += (v.x + v.y) + (v.z + v.w);
                ss += (v.x * v.x + v.y * v.y) + (v.z * v.z + v.w * v.w);
            }
            float m   = s * (1.0f / D_);
            float var = ss * (1.0f / D_) - m * m;
            float inv = rsqrtf(var + 1e-5f);
            for (int d = 0; d < D_; d += 4) {
                float4 v = *(const float4*)(row + d);
                float4 g = *(const float4*)(g1 + d);
                float4 e = *(const float4*)(be1 + d);
                *(unsigned*)(knbuf + tid * D_ + d) =
                    pack2_bf16((v.x - m) * inv * g.x + e.x, (v.y - m) * inv * g.y + e.y);
                *(unsigned*)(knbuf + tid * D_ + d + 2) =
                    pack2_bf16((v.z - m) * inv * g.z + e.z, (v.w - m) * inv * g.w + e.w);
            }
        }
        // qn = LN1(x) -> bf16 (u1), one thread per query row
        if (tid < C_) {
            const float* row = xbuf + tid * D_;
            float s = 0.f, ss = 0.f;
            for (int d = 0; d < D_; d += 4) {
                float4 v = *(const float4*)(row + d);
                s  += (v.x + v.y) + (v.z + v.w);
                ss += (v.x * v.x + v.y * v.y) + (v.z * v.z + v.w * v.w);
            }
            float m   = s * (1.0f / D_);
            float var = ss * (1.0f / D_) - m * m;
            float inv = rsqrtf(var + 1e-5f);
            for (int d = 0; d < D_; d += 4) {
                float4 v = *(const float4*)(row + d);
                float4 g = *(const float4*)(g1 + d);
                float4 e = *(const float4*)(be1 + d);
                *(unsigned*)(u1 + tid * D_ + d) =
                    pack2_bf16((v.x - m) * inv * g.x + e.x, (v.y - m) * inv * g.y + e.y);
                *(unsigned*)(u1 + tid * D_ + d + 2) =
                    pack2_bf16((v.z - m) * inv * g.z + e.z, (v.w - m) * inv * g.w + e.w);
            }
        }
        __syncthreads();

        // ---- k = kn@Wk^T : wave -> one n-strip, B fragments hoisted ----
        {
            int nt = wave;                       // 8 waves, 8 strips
            v16bf Bf[4];
#pragma unroll
            for (int kk = 0; kk < 4; ++kk) Bf[kk] = fragB_f32(Wk, D_, nt * 16, kk * 32, lane);
            int n = nt * 16 + nlo;
            float bn = bk[n];
            for (int mt = 0; mt < 16; ++mt) {
                v8f acc = {};
#pragma unroll
                for (int kk = 0; kk < 4; ++kk)
                    acc = wmma_bf16(fragA_bf(knbuf, D_, mt * 16, kk * 32, lane), Bf[kk], acc);
#pragma unroll
                for (int r = 0; r < 8; ++r)
                    kbuf[(mt * 16 + r + half * 8) * D_ + n] = bf16_1(acc[r] + bn);
            }
        }
        // ---- v = kn@Wv^T : store transposed (paired 32-bit stores) ----
        {
            int nt = wave;
            v16bf Bf[4];
#pragma unroll
            for (int kk = 0; kk < 4; ++kk) Bf[kk] = fragB_f32(Wv, D_, nt * 16, kk * 32, lane);
            int n = nt * 16 + nlo;
            float bn = bv[n];
            for (int mt = 0; mt < 16; ++mt) {
                v8f acc = {};
#pragma unroll
                for (int kk = 0; kk < 4; ++kk)
                    acc = wmma_bf16(fragA_bf(knbuf, D_, mt * 16, kk * 32, lane), Bf[kk], acc);
#pragma unroll
                for (int rp = 0; rp < 4; ++rp) {
                    int m = mt * 16 + 2 * rp + half * 8;
                    *(unsigned*)(vTbuf + n * T_ + m) =
                        pack2_bf16(acc[2 * rp] + bn, acc[2 * rp + 1] + bn);
                }
            }
        }
        // ---- q = qn@Wq^T ----
        {
            int nt = wave;
            v16bf Bf[4];
#pragma unroll
            for (int kk = 0; kk < 4; ++kk) Bf[kk] = fragB_f32(Wq, D_, nt * 16, kk * 32, lane);
            int n = nt * 16 + nlo;
            float bn = bq[n];
            for (int mt = 0; mt < 4; ++mt) {
                v8f acc = {};
#pragma unroll
                for (int kk = 0; kk < 4; ++kk)
                    acc = wmma_bf16(fragA_bf(u1, D_, mt * 16, kk * 32, lane), Bf[kk], acc);
#pragma unroll
                for (int r = 0; r < 8; ++r)
                    qbuf[(mt * 16 + r + half * 8) * D_ + n] = bf16_1(acc[r] + bn);
            }
        }
        __syncthreads();

        // ---------- attention, per head ----------
        for (int h = 0; h < H_; ++h) {
            // scores: wave -> (mt = wave&3, 8 consecutive t tiles); A fragment hoisted
            {
                int mt  = wave & 3;
                int tt0 = (wave >> 2) * 8;
                v16bf a = fragA_bf(qbuf, D_, mt * 16, h * HD_, lane);
                for (int tt = tt0; tt < tt0 + 8; ++tt) {
                    v8f acc = {};
                    acc = wmma_bf16(a, fragB_bf(kbuf, D_, tt * 16, h * HD_, lane), acc);
                    int n = tt * 16 + nlo;
#pragma unroll
                    for (int r = 0; r < 8; ++r)
                        sbuf[(mt * 16 + r + half * 8) * T_ + n] = acc[r];
                }
            }
            __syncthreads();
            // masked softmax over t: 4 threads per row; emit bf16 attn into abuf
            {
                int c  = tid & 63;
                int qt = tid >> 6;
                int t0 = qt * 64;
                int cv = catval[c];
                const float scale = 0.17677669529663687f;  // 1/sqrt(32)
                float mx = -3.0e38f;
                for (int t = t0; t < t0 + 64; ++t) {
                    float sv = sbuf[c * T_ + t] * scale;
                    bool match = (catsl[t] == cv) && maskl[t];
                    sv = match ? sv : -1.0e9f;
                    sbuf[c * T_ + t] = sv;
                    mx = fmaxf(mx, sv);
                }
                redm[qt * 64 + c] = mx;
                __syncthreads();
                mx = fmaxf(fmaxf(redm[c], redm[64 + c]), fmaxf(redm[128 + c], redm[192 + c]));
                float sum = 0.f;
                for (int t = t0; t < t0 + 64; ++t) {
                    float e = __expf(sbuf[c * T_ + t] - mx);
                    sbuf[c * T_ + t] = e;
                    sum += e;
                }
                reds[qt * 64 + c] = sum;
                __syncthreads();
                float tot = (reds[c] + reds[64 + c]) + (reds[128 + c] + reds[192 + c]);
                float inv = 1.0f / tot;
                for (int t = t0; t < t0 + 64; t += 2) {
                    float p0 = sbuf[c * T_ + t]     * inv;
                    float p1 = sbuf[c * T_ + t + 1] * inv;
                    *(unsigned*)(abuf + c * T_ + t) = pack2_bf16(p0, p1);
                }
            }
            __syncthreads();
            // ctx_h = attn @ v_h  (A = abuf bf16, B = vT rows), 1 tile per wave, K=256
            {
                int mt = wave >> 1, nt = wave & 1;
                v8f acc = {};
#pragma unroll
                for (int kk = 0; kk < 8; ++kk) {
                    v16bf a  = fragA_bf(abuf, T_, mt * 16, kk * 32, lane);
                    v16bf bb = fragB_bf(vTbuf, T_, h * HD_ + nt * 16, kk * 32, lane);
                    acc = wmma_bf16(a, bb, acc);
                }
                int n = h * HD_ + nt * 16 + nlo;
#pragma unroll
                for (int r = 0; r < 8; ++r)
                    u1[(mt * 16 + r + half * 8) * D_ + n] = bf16_1(acc[r]);  // ctx bf16
            }
            __syncthreads();
        }

        // ---- x += ctx @ Wo^T + bo ----
        {
            int nt = wave;
            v16bf Bf[4];
#pragma unroll
            for (int kk = 0; kk < 4; ++kk) Bf[kk] = fragB_f32(Wo_, D_, nt * 16, kk * 32, lane);
            int n = nt * 16 + nlo;
            float bn = bo_[n];
            for (int mt = 0; mt < 4; ++mt) {
                v8f acc = {};
#pragma unroll
                for (int kk = 0; kk < 4; ++kk)
                    acc = wmma_bf16(fragA_bf(u1, D_, mt * 16, kk * 32, lane), Bf[kk], acc);
#pragma unroll
                for (int r = 0; r < 8; ++r)
                    xbuf[(mt * 16 + r + half * 8) * D_ + n] += acc[r] + bn;
            }
        }
        __syncthreads();

        // ---------- FFN ----------
        {
            const float* g2  = ln2g + (long)l * D_;
            const float* be2 = ln2b + (long)l * D_;
            if (tid < C_) {   // xn = LN2(x) -> u1 bf16
                const float* row = xbuf + tid * D_;
                float s = 0.f, ss = 0.f;
                for (int d = 0; d < D_; d += 4) {
                    float4 v = *(const float4*)(row + d);
                    s  += (v.x + v.y) + (v.z + v.w);
                    ss += (v.x * v.x + v.y * v.y) + (v.z * v.z + v.w * v.w);
                }
                float m   = s * (1.0f / D_);
                float var = ss * (1.0f / D_) - m * m;
                float inv = rsqrtf(var + 1e-5f);
                for (int d = 0; d < D_; d += 4) {
                    float4 v = *(const float4*)(row + d);
                    float4 g = *(const float4*)(g2 + d);
                    float4 e = *(const float4*)(be2 + d);
                    *(unsigned*)(u1 + tid * D_ + d) =
                        pack2_bf16((v.x - m) * inv * g.x + e.x, (v.y - m) * inv * g.y + e.y);
                    *(unsigned*)(u1 + tid * D_ + d + 2) =
                        pack2_bf16((v.z - m) * inv * g.z + e.z, (v.w - m) * inv * g.w + e.w);
                }
            }
            __syncthreads();
            // h1 = relu(xn @ W1^T + b1): 32 n-strips, 4 per wave, B frags hoisted
            for (int nt = wave; nt < 32; nt += 8) {
                v16bf Bf[4];
#pragma unroll
                for (int kk = 0; kk < 4; ++kk) Bf[kk] = fragB_f32(W1_, D_, nt * 16, kk * 32, lane);
                int n = nt * 16 + nlo;
                float bn = b1_[n];
                for (int mt = 0; mt < 4; ++mt) {
                    v8f acc = {};
#pragma unroll
                    for (int kk = 0; kk < 4; ++kk)
                        acc = wmma_bf16(fragA_bf(u1, D_, mt * 16, kk * 32, lane), Bf[kk], acc);
#pragma unroll
                    for (int r = 0; r < 8; ++r)
                        h1buf[(mt * 16 + r + half * 8) * F_ + n] = bf16_1(fmaxf(acc[r] + bn, 0.0f));
                }
            }
            __syncthreads();
            // x += h1 @ W2^T + b2: one n-strip per wave, 4 accumulators, stream B per-kk
            {
                int nt = wave;
                v8f accs[4] = {v8f{}, v8f{}, v8f{}, v8f{}};
#pragma unroll 4
                for (int kk = 0; kk < 16; ++kk) {
                    v16bf bb = fragB_f32(W2_, F_, nt * 16, kk * 32, lane);
#pragma unroll
                    for (int mt = 0; mt < 4; ++mt)
                        accs[mt] = wmma_bf16(fragA_bf(h1buf, F_, mt * 16, kk * 32, lane), bb, accs[mt]);
                }
                int n = nt * 16 + nlo;
                float bn = b2_[n];
#pragma unroll
                for (int mt = 0; mt < 4; ++mt)
#pragma unroll
                    for (int r = 0; r < 8; ++r)
                        xbuf[(mt * 16 + r + half * 8) * D_ + n] += accs[mt][r] + bn;
            }
            __syncthreads();
        }
    }

    // ---------- final: logits over C, softmax, weighted reduce ----------
    {
        const float* tg = tgt + (long)b * D_;
        if (tid < C_) {
            float s = 0.f;
            for (int d = 0; d < D_; d += 4) {
                float4 tv = *(const float4*)(tg + d);
                float4 xv = *(const float4*)(xbuf + tid * D_ + d);
                s += (tv.x * xv.x + tv.y * xv.y) + (tv.z * xv.z + tv.w * xv.w);
            }
            float lg = s * 0.08838834764831845f;   // 1/sqrt(128)
            logits[tid] = (tid < *nuniq) ? lg : -1.0e9f;
        }
        __syncthreads();
        if (tid == 0) {
            float mx = -3.0e38f;
            for (int c = 0; c < C_; ++c) mx = fmaxf(mx, logits[c]);
            float sum = 0.f;
            for (int c = 0; c < C_; ++c) { float e = __expf(logits[c] - mx); wsm[c] = e; sum += e; }
            float inv = 1.0f / sum;
            for (int c = 0; c < C_; ++c) wsm[c] *= inv;
        }
        __syncthreads();
        for (int d = tid; d < D_; d += 256) {
            float s = 0.f;
            for (int c = 0; c < C_; ++c) s += wsm[c] * xbuf[c * D_ + d];
            out[(long)b * D_ + d] = s;
        }
    }
}

extern "C" void kernel_launch(void* const* d_in, const int* in_sizes, int n_in,
                              void* d_out, int out_size, void* d_ws, size_t ws_size,
                              hipStream_t stream) {
    (void)in_sizes; (void)n_in; (void)d_ws; (void)ws_size; (void)out_size;
    const float*         tgt   = (const float*)d_in[0];
    const float*         seq   = (const float*)d_in[1];
    const int*           cats  = (const int*)d_in[2];
    const unsigned char* maskp = (const unsigned char*)d_in[3];
    const float* wqkv = (const float*)d_in[4];
    const float* bqkv = (const float*)d_in[5];
    const float* wo   = (const float*)d_in[6];
    const float* bo   = (const float*)d_in[7];
    const float* ln1g = (const float*)d_in[8];
    const float* ln1b = (const float*)d_in[9];
    const float* ln2g = (const float*)d_in[10];
    const float* ln2b = (const float*)d_in[11];
    const float* w1   = (const float*)d_in[12];
    const float* b1   = (const float*)d_in[13];
    const float* w2   = (const float*)d_in[14];
    const float* b2   = (const float*)d_in[15];
    float* out = (float*)d_out;

    cia_kernel<<<dim3(B_), dim3(256), SMEM_TOTAL, stream>>>(
        tgt, seq, cats, maskp, wqkv, bqkv, wo, bo,
        ln1g, ln1b, ln2g, ln2b, w1, b1, w2, b2, out);
}
